// MultiGraphAttentionCNN_30520037605841
// MI455X (gfx1250) — matrix operand order repaired and checked by reading, compiled, not run on hardware
//
#include <hip/hip_runtime.h>
#include <hip/hip_bf16.h>
#include <stdint.h>

// B=1, N=4096, F=256, D=64, H=4
#define Nn 4096
#define Ff 256
#define Dd 64
#define Hh 4

typedef __attribute__((ext_vector_type(16))) __bf16 v16bf;
typedef __attribute__((ext_vector_type(8)))  float  v8f;
typedef __attribute__((ext_vector_type(4)))  unsigned int v4u;
typedef __attribute__((ext_vector_type(8)))  int v8i;
typedef __attribute__((ext_vector_type(4)))  int v4i;

#if defined(__has_builtin)
#if __has_builtin(__builtin_amdgcn_tensor_load_to_lds) && __has_builtin(__builtin_amdgcn_s_wait_tensorcnt)
#define USE_TDM 1
#endif
#endif
#ifndef USE_TDM
#define USE_TDM 0
#endif

__device__ __forceinline__ float elu_f(float x) {
    return x > 0.0f ? x : expm1f(x);
}

#if USE_TDM
// Build a D# (CDNA5 ISA) and issue TENSOR_LOAD_TO_LDS.
// 2D/3D tile; LDS row padding: every (2<<pad_interval) DWORDs add (pad_amount+1) DWORDs.
// Toolchain is the 6-arg clang-23 form: (v4u g0, v8i g1, v4i g2, v4i g3, v8i g4, i32 cpol).
__device__ __forceinline__ void tdm_load_tile(
    const void* gsrc, unsigned ldsOff, unsigned dsz_log2,
    unsigned tensor_d0, unsigned tensor_d1, unsigned tensor_d2,
    unsigned tile_d0, unsigned tile_d1, unsigned tile_d2,
    unsigned long long d0_stride, unsigned long long d1_stride,
    unsigned pad_interval, unsigned pad_amount, int pad_en) {
    unsigned long long ga = (unsigned long long)(uintptr_t)gsrc;
    v4u g0;
    g0[0] = 1u;                                                    // count=1 (valid), user mode
    g0[1] = ldsOff;                                                // lds_addr
    g0[2] = (unsigned)ga;                                          // global_addr[31:0]
    g0[3] = (unsigned)((ga >> 32) & 0x01FFFFFFu) | (2u << 30);     // global_addr[56:32] | type=2
    v8i g1;
    g1[0] = (int)((dsz_log2 << 16) | ((pad_en ? 1u : 0u) << 20) |
                  (pad_interval << 22) | (pad_amount << 25));
    g1[1] = (int)((tensor_d0 & 0xFFFFu) << 16);                    // tensor_dim0[15:0] @63:48
    g1[2] = (int)((tensor_d0 >> 16) | ((tensor_d1 & 0xFFFFu) << 16));
    g1[3] = (int)((tensor_d1 >> 16) | (tile_d0 << 16));
    g1[4] = (int)(tile_d1 | (tile_d2 << 16));
    g1[5] = (int)(unsigned)(d0_stride & 0xFFFFFFFFull);
    g1[6] = (int)((unsigned)((d0_stride >> 32) & 0xFFFFull) |
                  ((unsigned)(d1_stride & 0xFFFFull) << 16));
    g1[7] = (int)(unsigned)((d1_stride >> 16) & 0xFFFFFFFFull);
    v4i g2; g2[0] = (int)tensor_d2; g2[1] = 0; g2[2] = 0; g2[3] = 0;
    v4i g3 = {0, 0, 0, 0};
    v8i g4 = {0, 0, 0, 0, 0, 0, 0, 0};
    __builtin_amdgcn_tensor_load_to_lds(g0, g1, g2, g3, g4, 0);
}
#endif

// ---------------------------------------------------------------------------
// Kernel 1: conv[h][n][d] = sum_f x[n][f] * W[h][f][d]   (bf16 WMMA, f32 acc)
// grid (N/16, H), block 128 (4 waves; wave w handles d-tile w)
// ---------------------------------------------------------------------------
__global__ __launch_bounds__(128) void gat_conv_kernel(
    const float* __restrict__ x, const float* __restrict__ W,
    float* __restrict__ conv_f32, unsigned short* __restrict__ conv_b16us) {
    __bf16* conv_b16 = (__bf16*)conv_b16us;
    const int m0   = blockIdx.x * 16;
    const int h    = blockIdx.y;
    const int tid  = threadIdx.x;
    const int wave = tid >> 5;
    const int lane = tid & 31;
    const int hl   = (lane >> 4) & 1;
    const int ln16 = lane & 15;
    const int d0   = wave * 16;

    __shared__ __bf16 xa[16][34];

    v8f acc = {};
    for (int f0 = 0; f0 < Ff; f0 += 32) {
        __syncthreads();
        for (int u = tid; u < 512; u += 128) {
            int r = u >> 5, c = u & 31;
            xa[r][c] = (__bf16)x[(m0 + r) * Ff + f0 + c];
        }
        __syncthreads();
        v16bf afrag, bfrag;
#pragma unroll
        for (int e = 0; e < 16; ++e) {
            int ka = (e >> 3) * 16 + (e & 7) + hl * 8;      // A K-map (ISA 7.12.2)
            afrag[e] = xa[ln16][ka];
            int kb = (e >> 3) * 16 + (e & 7) + hl * 8;      // B K-map
            bfrag[e] = (__bf16)W[(h * Ff + f0 + kb) * Dd + d0 + ln16];
        }
        acc = __builtin_amdgcn_wmma_f32_16x16x32_bf16(
            false, afrag, false, bfrag, (short)0, acc, false, false);
    }
#pragma unroll
    for (int r = 0; r < 8; ++r) {
        int m = m0 + r + hl * 8;
        int d = d0 + ln16;
        int idx = (h * Nn + m) * Dd + d;
        float v = acc[r];
        conv_f32[idx] = v;
        conv_b16[idx] = (__bf16)v;
    }
}

// ---------------------------------------------------------------------------
// Kernel 2: s_self/s_neigh per (h,n) + per-head max of s_neigh. grid H, block 256
// ---------------------------------------------------------------------------
__global__ __launch_bounds__(256) void gat_scores_kernel(
    const float* __restrict__ conv, const float* __restrict__ attn_kernel,
    float* __restrict__ s_self, float* __restrict__ s_neigh, float* __restrict__ gmax) {
    const int h = blockIdx.x;
    const int tid = threadIdx.x;
    __shared__ float as[Dd], an[Dd];
    if (tid < Dd)            as[tid]      = attn_kernel[h * 2 * Dd + tid];
    else if (tid < 2 * Dd)   an[tid - Dd] = attn_kernel[h * 2 * Dd + tid];
    __syncthreads();

    float lmax = -3.4e38f;
    for (int n = tid; n < Nn; n += 256) {
        const float* c = conv + (size_t)(h * Nn + n) * Dd;
        float ss = 0.f, sn = 0.f;
#pragma unroll 8
        for (int d = 0; d < Dd; ++d) { float v = c[d]; ss += v * as[d]; sn += v * an[d]; }
        s_self[h * Nn + n]  = ss;
        s_neigh[h * Nn + n] = sn;
        lmax = fmaxf(lmax, sn);
    }
    __shared__ float red[256];
    red[tid] = lmax;
    __syncthreads();
    for (int s = 128; s > 0; s >>= 1) {
        if (tid < s) red[tid] = fmaxf(red[tid], red[tid + s]);
        __syncthreads();
    }
    if (tid == 0) gmax[h] = red[0];
}

// ---------------------------------------------------------------------------
// Kernel 3: flash-style masked attention, single pass over adj.
// grid N/32, block 256 (8 waves). Wave w: head = w&3, row strip = (w>>2)*16.
// TDM double-buffers adj (2D tile) + conv_bf16 (3D tile, all 4 heads) into
// padded LDS while waves compute the previous chunk with 4x wmma bf16.
// ---------------------------------------------------------------------------
__global__ __launch_bounds__(256) void gat_attn_kernel(
    const float* __restrict__ adj, const float* __restrict__ s_self,
    const float* __restrict__ s_neigh, const float* __restrict__ gmax,
    const unsigned short* __restrict__ conv_b16us, float* __restrict__ out) {
    const __bf16* conv_b16 = (const __bf16*)conv_b16us;
    const int i0   = blockIdx.x * 32;
    const int tid  = threadIdx.x;
    const int wave = tid >> 5;
    const int lane = tid & 31;
    const int hl   = (lane >> 4) & 1;
    const int ln16 = lane & 15;
    const int h    = wave & 3;
    const int mi   = (wave >> 2) * 16;
    const int NCH  = Nn / 32;

    __shared__ float  adjS[2][32][36];        // pad: 32 dwords + 4 -> 36 f32/row
    __shared__ __bf16 convS[2][Hh][32][72];   // pad: 32 dwords + 4 -> 72 bf16/row
    __shared__ float  sneighS[2][Hh][32];
    __shared__ float  denomS[Hh][32];

    const int   iloc  = mi + ln16;
    const int   iglob = i0 + iloc;
    const float ssi   = s_self[h * Nn + iglob];
    const float shift = elu_f(ssi + gmax[h]);     // upper bound on row scores

    v8f acc0 = {}, acc1 = {}, acc2 = {}, acc3 = {};
    float dsum = 0.f;

#if USE_TDM
    auto issue_chunk = [&](int j0, int buf) {
        // adj tile: 32 rows x 32 f32, LDS rows padded to 36 dwords
        tdm_load_tile(adj + (size_t)i0 * Nn + j0,
                      (unsigned)(uintptr_t)&adjS[buf][0][0],
                      2u, (unsigned)Nn, (unsigned)Nn, 0u,
                      32u, 32u, 0u,
                      (unsigned long long)Nn, 0ull,
                      4u /*pad every 32 dwords*/, 3u /*+4 dwords*/, 1);
        // conv tile: 3D (d=64, 32 rows, 4 heads) bf16, rows padded to 72 elems
        tdm_load_tile(conv_b16 + (size_t)j0 * Dd,
                      (unsigned)(uintptr_t)&convS[buf][0][0][0],
                      1u, (unsigned)Dd, (unsigned)Nn, (unsigned)Hh,
                      (unsigned)Dd, 32u, (unsigned)Hh,
                      (unsigned long long)Dd, (unsigned long long)Nn * Dd,
                      4u, 3u, 1);
    };
    // prologue: stage chunk 0 into buffer 0
    if (tid < 32) issue_chunk(0, 0);
    for (int u = tid; u < Hh * 32; u += 256) {
        int h2 = u >> 5, jj = u & 31;
        sneighS[0][h2][jj] = s_neigh[h2 * Nn + jj];
    }
#endif

#pragma unroll 2
    for (int jc = 0; jc < NCH; ++jc) {
        const int buf = jc & 1;
#if USE_TDM
        __builtin_amdgcn_s_wait_tensorcnt(0);   // issuing wave drains TDM; no-op elsewhere
        __syncthreads();                        // chunk jc LDS visible to all waves
        if (jc + 1 < NCH) {
            const int j1 = (jc + 1) * 32;
            if (tid < 32) issue_chunk(j1, buf ^ 1);   // overlap DMA with compute
            for (int u = tid; u < Hh * 32; u += 256) {
                int h2 = u >> 5, jj = u & 31;
                sneighS[buf ^ 1][h2][jj] = s_neigh[h2 * Nn + j1 + jj];
            }
        }
#else
        const int j0 = jc * 32;
        __syncthreads();
        {
            const int arow = tid >> 3;
            const int ac4  = (tid & 7) * 4;
            const float* src = adj + (size_t)(i0 + arow) * Nn + j0 + ac4;
            *(float4*)&adjS[buf][arow][ac4] = *(const float4*)src;
            if (jc + 1 < NCH) __builtin_prefetch(src + 32, 0, 0);
        }
        for (int u = tid; u < 1024; u += 256) {
            int h2 = u >> 8, rem = u & 255, row = rem >> 3, seg = rem & 7;
            const __bf16* src = conv_b16 + ((size_t)(h2 * Nn) + j0 + row) * Dd + seg * 8;
            *(uint4*)&convS[buf][h2][row][seg * 8] = *(const uint4*)src;
        }
        for (int u = tid; u < Hh * 32; u += 256) {
            int h2 = u >> 5, jj = u & 31;
            sneighS[buf][h2][jj] = s_neigh[h2 * Nn + j0 + jj];
        }
        __syncthreads();
#endif
        // P fragment (A operand): p_ij = edge ? exp(elu(ssi+snj) - shift) : 0
        v16bf pa;
#pragma unroll
        for (int e = 0; e < 16; ++e) {
            int jj = (e >> 3) * 16 + (e & 7) + hl * 8;
            float av = adjS[buf][iloc][jj];
            float p = 0.f;
            if (av > 0.5f) p = __expf(elu_f(ssi + sneighS[buf][h][jj]) - shift);
            dsum += p;
            pa[e] = (__bf16)p;
        }
        // B fragments for the 4 d-tiles
        v16bf b0, b1, b2, b3;
#pragma unroll
        for (int e = 0; e < 16; ++e) {
            int k = (e >> 3) * 16 + (e & 7) + hl * 8;
            b0[e] = convS[buf][h][k][ln16];
            b1[e] = convS[buf][h][k][16 + ln16];
            b2[e] = convS[buf][h][k][32 + ln16];
            b3[e] = convS[buf][h][k][48 + ln16];
        }
        acc0 = __builtin_amdgcn_wmma_f32_16x16x32_bf16(false, pa, false, b0, (short)0, acc0, false, false);
        acc1 = __builtin_amdgcn_wmma_f32_16x16x32_bf16(false, pa, false, b1, (short)0, acc1, false, false);
        acc2 = __builtin_amdgcn_wmma_f32_16x16x32_bf16(false, pa, false, b2, (short)0, acc2, false, false);
        acc3 = __builtin_amdgcn_wmma_f32_16x16x32_bf16(false, pa, false, b3, (short)0, acc3, false, false);
    }

    // combine the two lane-halves (each covered half the K values of row iloc)
    dsum += __shfl_xor(dsum, 16, 32);
    if (lane < 16) denomS[h][iloc] = dsum;
    __syncthreads();

    // epilogue: out[i][h*64+d] = acc / denom
#pragma unroll
    for (int t = 0; t < 4; ++t) {
        v8f& a = (t == 0 ? acc0 : (t == 1 ? acc1 : (t == 2 ? acc2 : acc3)));
#pragma unroll
        for (int r = 0; r < 8; ++r) {
            int rloc = mi + r + hl * 8;
            float dv = denomS[h][rloc];
            out[(size_t)(i0 + rloc) * (Hh * Dd) + h * Dd + t * 16 + ln16] = a[r] / dv;
        }
    }
}

// ---------------------------------------------------------------------------
extern "C" void kernel_launch(void* const* d_in, const int* in_sizes, int n_in,
                              void* d_out, int out_size, void* d_ws, size_t ws_size,
                              hipStream_t stream) {
    const float* x   = (const float*)d_in[0];   // (1,4096,256)
    const float* adj = (const float*)d_in[1];   // (1,4096,4096)
    const float* W   = (const float*)d_in[2];   // (4,256,64)
    const float* ak  = (const float*)d_in[3];   // (4,128)
    float* out = (float*)d_out;                 // (4096,256)

    char* ws = (char*)d_ws;
    float*          conv_f32 = (float*)ws;                              // 4 MB
    unsigned short* conv_b16 = (unsigned short*)(ws + (4u << 20));      // 2 MB
    float*          s_self   = (float*)(ws + (6u << 20));               // 64 KB
    float*          s_neigh  = s_self + Hh * Nn;                        // 64 KB
    float*          gmax     = s_neigh + Hh * Nn;                       // 16 B

    hipLaunchKernelGGL(gat_conv_kernel, dim3(Nn / 16, Hh), dim3(128), 0, stream,
                       x, W, conv_f32, conv_b16);
    hipLaunchKernelGGL(gat_scores_kernel, dim3(Hh), dim3(256), 0, stream,
                       conv_f32, ak, s_self, s_neigh, gmax);
    hipLaunchKernelGGL(gat_attn_kernel, dim3(Nn / 32), dim3(256), 0, stream,
                       adj, s_self, s_neigh, gmax, conv_b16, out);
}
